// _RealNVPCoupling_16870631539268
// MI455X (gfx1250) — compile-verified, matrix operand order checked by reading, not run on hardware
//
#include <hip/hip_runtime.h>

// ---------------------------------------------------------------------------
// RealNVP coupling layer, fused bf16-WMMA implementation for gfx1250 (MI455X).
//
// Shapes: B=65536 rows, x is (B,256). cond = x[:, :128], x_t = x[:, 128:].
// Two MLPs 128 -> 512 -> 512 -> 128 (ReLU, ReLU, linear) on cond.
// y[:, :128] = cond ; y[:, 128:] = x_t * exp(tanh(mlp_s)) + mlp_t
// log_det = sum(tanh(mlp_s), axis=1)
// d_out = [ y (B*256 f32) | log_det (B f32) ]
// ---------------------------------------------------------------------------

typedef __attribute__((ext_vector_type(16))) __bf16 v16bf;
typedef __attribute__((ext_vector_type(8)))  __bf16 v8bf;
typedef __attribute__((ext_vector_type(8)))  float  v8f;

#define NROWS   65536
#define DIM     256
#define CONDW   128
#define HID     512
#define TOUT    128

// padded LDS strides (in halves) to spread rows across banks
#define H1_STRIDE 520   // 512 + 8
#define H2_STRIDE 72    // 64 + 8

// ---------------------------------------------------------------------------
// Weight repack: f32 row-major (K,N) -> bf16 WMMA B-fragment order.
// Tile (nt,kt) covers W[kt*32 .. kt*32+31, nt*16 .. nt*16+15], stored as 512
// contiguous bf16: flat = (nt*(K/32)+kt)*512 + lane*16 + e, where
//   n = nt*16 + (lane & 15)
//   k = kt*32 + e + (e>=8 ? 8 : 0) + (lane>=16 ? 8 : 0)
// matching the ISA 16-bit 32x16 B-matrix VGPR layout (wave32).
// ---------------------------------------------------------------------------
__global__ void pack_w_kernel(const float* __restrict__ W,
                              __bf16* __restrict__ out, int K, int N) {
    int id = blockIdx.x * 256 + threadIdx.x;
    int total = K * N;
    if (id >= total) return;
    int tile   = id >> 9;          // / 512
    int within = id & 511;
    int lane   = within >> 4;
    int e      = within & 15;
    int ktiles = K >> 5;
    int nt = tile / ktiles;
    int kt = tile - nt * ktiles;
    int n = (nt << 4) + (lane & 15);
    int k = (kt << 5) + e + ((e >= 8) ? 8 : 0) + ((lane >= 16) ? 8 : 0);
    out[id] = (__bf16)W[(size_t)k * N + n];
}

// ---------------------------------------------------------------------------
// Fragment helpers
// ---------------------------------------------------------------------------
__device__ __forceinline__ v8f wmma_bf16(v16bf a, v16bf b, v8f c) {
    // (neg_a, A, neg_b, B, c_mod, C, reuse_a, reuse_b)
    return __builtin_amdgcn_wmma_f32_16x16x32_bf16(
        false, a, false, b, (short)0, c, false, false);
}

// B fragment: one v16bf per lane, contiguous 32B (2 x global_load_b128)
__device__ __forceinline__ v16bf load_wfrag(const __bf16* __restrict__ Wp,
                                            int ktiles, int nt, int kt, int lane) {
    const v16bf* p = (const v16bf*)(Wp + (((size_t)nt * ktiles + kt) << 9));
    return p[lane];
}

// A fragment from bf16 row-major LDS (per-wave region), rows 0..15,
// per-lane K pattern = two contiguous 8-half runs (2 x ds_load_b128)
__device__ __forceinline__ v16bf frag_a_lds(const __bf16* base, int strideH,
                                            int kt, int lane) {
    int row = lane & 15;
    int kb  = (lane >= 16) ? 8 : 0;
    const __bf16* r = base + row * strideH + (kt << 5) + kb;
    v8bf lo = *(const v8bf*)(r);
    v8bf hi = *(const v8bf*)(r + 16);
    v16bf a;
#pragma unroll
    for (int i = 0; i < 8; i++) { a[i] = lo[i]; a[8 + i] = hi[i]; }
    return a;
}

// A fragment straight from global f32 (cond rows), converted to bf16
__device__ __forceinline__ v16bf frag_a_gf32(const float* __restrict__ rowp,
                                             int kt, int lane) {
    const float* r = rowp + (kt << 5) + ((lane >= 16) ? 8 : 0);
    v16bf a;
#pragma unroll
    for (int i = 0; i < 8; i++)  a[i]     = (__bf16)r[i];
#pragma unroll
    for (int i = 0; i < 8; i++)  a[8 + i] = (__bf16)r[16 + i];
    return a;
}

// ---------------------------------------------------------------------------
// One MLP (128 -> 512 -> 512 -> 128) for this wave's 16 rows.
// Output (with bias, pre-nonlinearity) left in acc3[8] (16x128 C-layout).
// h1  : per-wave LDS, 16 x HID bf16 (stride H1_STRIDE)
// h2c : per-wave LDS, 16 x 64  bf16 (stride H2_STRIDE), chunk staging
// ---------------------------------------------------------------------------
__device__ __forceinline__ void mlp_eval(
    const float* __restrict__ x, int rowBase, int lane,
    const __bf16* __restrict__ W1p, const float* __restrict__ b1,
    const __bf16* __restrict__ W2p, const float* __restrict__ b2,
    const __bf16* __restrict__ W3p, const float* __restrict__ b3,
    __bf16* h1, __bf16* h2c, v8f acc3[8]) {

    const int col   = lane & 15;
    const int rhalf = (lane >= 16) ? 8 : 0;
    const float* xrow = x + (size_t)(rowBase + (lane & 15)) * DIM;  // cond cols

    // ---- GEMM1: h1 = relu(cond @ W1 + b1), 16x512 ----
    v16bf ca[4];
#pragma unroll
    for (int kt = 0; kt < 4; kt++) ca[kt] = frag_a_gf32(xrow, kt, lane);

    for (int nt = 0; nt < 32; nt++) {
        v8f acc;
#pragma unroll
        for (int r = 0; r < 8; r++) acc[r] = 0.0f;
#pragma unroll
        for (int kt = 0; kt < 4; kt++) {
            v16bf b = load_wfrag(W1p, 4, nt, kt, lane);
            acc = wmma_bf16(ca[kt], b, acc);
        }
        float bias = b1[nt * 16 + col];
#pragma unroll
        for (int r = 0; r < 8; r++) {
            float v = acc[r] + bias;
            v = v > 0.0f ? v : 0.0f;
            h1[(r + rhalf) * H1_STRIDE + nt * 16 + col] = (__bf16)v;
        }
    }

    // ---- GEMM2 (chunked) fused with GEMM3 accumulation ----
#pragma unroll
    for (int nt = 0; nt < 8; nt++)
#pragma unroll
        for (int r = 0; r < 8; r++) acc3[nt][r] = 0.0f;

    for (int c = 0; c < 8; c++) {           // 8 chunks of 64 h2-columns
        v8f acc2[4];
#pragma unroll
        for (int j = 0; j < 4; j++)
#pragma unroll
            for (int r = 0; r < 8; r++) acc2[j][r] = 0.0f;

        for (int kt = 0; kt < 16; kt++) {   // K = 512
            v16bf a = frag_a_lds(h1, H1_STRIDE, kt, lane);
#pragma unroll
            for (int j = 0; j < 4; j++) {
                v16bf b = load_wfrag(W2p, 16, c * 4 + j, kt, lane);
                acc2[j] = wmma_bf16(a, b, acc2[j]);
            }
        }
        // relu+bias -> h2 chunk staging (16 x 64)
#pragma unroll
        for (int j = 0; j < 4; j++) {
            float bias = b2[c * 64 + j * 16 + col];
#pragma unroll
            for (int r = 0; r < 8; r++) {
                float v = acc2[j][r] + bias;
                v = v > 0.0f ? v : 0.0f;
                h2c[(r + rhalf) * H2_STRIDE + j * 16 + col] = (__bf16)v;
            }
        }
        // consume chunk into GEMM3 accumulators (N = 128)
#pragma unroll
        for (int ktl = 0; ktl < 2; ktl++) {
            v16bf a = frag_a_lds(h2c, H2_STRIDE, ktl, lane);
            int ktg = c * 2 + ktl;
#pragma unroll
            for (int nt = 0; nt < 8; nt++) {
                v16bf b = load_wfrag(W3p, 16, nt, ktg, lane);
                acc3[nt] = wmma_bf16(a, b, acc3[nt]);
            }
        }
    }

    // bias3
#pragma unroll
    for (int nt = 0; nt < 8; nt++) {
        float bias = b3[nt * 16 + col];
#pragma unroll
        for (int r = 0; r < 8; r++) acc3[nt][r] += bias;
    }
}

// ---------------------------------------------------------------------------
// Main fused kernel: 512 WGs x 256 threads (8 waves, 16 rows/wave)
// ---------------------------------------------------------------------------
__global__ void __launch_bounds__(256)
realnvp_kernel(const float* __restrict__ x,
               const float* __restrict__ bs1, const float* __restrict__ bs2,
               const float* __restrict__ bs3,
               const float* __restrict__ bt1, const float* __restrict__ bt2,
               const float* __restrict__ bt3,
               const __bf16* __restrict__ Wpack,
               float* __restrict__ y, float* __restrict__ logdet) {
    extern __shared__ __bf16 lds[];
    const int tid  = threadIdx.x;
    const int wave = tid >> 5;
    const int lane = tid & 31;
    const int wgRow   = blockIdx.x * 128;
    const int rowBase = wgRow + wave * 16;

    // passthrough copy: y[:, :128] = cond   (coalesced)
    for (int i = tid; i < 128 * CONDW; i += 256) {
        int r = i >> 7, cc = i & 127;
        size_t idx = (size_t)(wgRow + r) * DIM + cc;
        y[idx] = x[idx];
    }

    __bf16* h1  = lds + wave * (16 * H1_STRIDE);
    __bf16* h2c = lds + 8 * (16 * H1_STRIDE) + wave * (16 * H2_STRIDE);

    const __bf16* Ws1p = Wpack;
    const __bf16* Ws2p = Ws1p + CONDW * HID;
    const __bf16* Ws3p = Ws2p + HID * HID;
    const __bf16* Wt1p = Ws3p + HID * TOUT;
    const __bf16* Wt2p = Wt1p + CONDW * HID;
    const __bf16* Wt3p = Wt2p + HID * HID;

    // ---- s MLP, tanh, logdet ----
    v8f sacc[8];
    mlp_eval(x, rowBase, lane, Ws1p, bs1, Ws2p, bs2, Ws3p, bs3, h1, h2c, sacc);

#pragma unroll
    for (int nt = 0; nt < 8; nt++)
#pragma unroll
        for (int r = 0; r < 8; r++) sacc[nt][r] = tanhf(sacc[nt][r]);

    const int rhalf = (lane >= 16) ? 8 : 0;
#pragma unroll
    for (int r = 0; r < 8; r++) {
        float p = 0.0f;
#pragma unroll
        for (int nt = 0; nt < 8; nt++) p += sacc[nt][r];
        // reduce across the 16-lane half (wave32): masks stay inside the half
        p += __shfl_xor(p, 1, 32);
        p += __shfl_xor(p, 2, 32);
        p += __shfl_xor(p, 4, 32);
        p += __shfl_xor(p, 8, 32);
        if ((lane & 15) == 0) logdet[rowBase + r + rhalf] = p;
    }

    // ---- t MLP + fused affine tail ----
    v8f tacc[8];
    mlp_eval(x, rowBase, lane, Wt1p, bt1, Wt2p, bt2, Wt3p, bt3, h1, h2c, tacc);

    const int col = lane & 15;
#pragma unroll
    for (int nt = 0; nt < 8; nt++) {
#pragma unroll
        for (int r = 0; r < 8; r++) {
            size_t idx = (size_t)(rowBase + r + rhalf) * DIM + CONDW + nt * 16 + col;
            float xv = x[idx];
            y[idx] = xv * __expf(sacc[nt][r]) + tacc[nt][r];
        }
    }
}

// ---------------------------------------------------------------------------
// Host launcher
// ---------------------------------------------------------------------------
extern "C" void kernel_launch(void* const* d_in, const int* in_sizes, int n_in,
                              void* d_out, int out_size, void* d_ws, size_t ws_size,
                              hipStream_t stream) {
    const float* x   = (const float*)d_in[0];
    const float* Ws1 = (const float*)d_in[1];
    const float* bs1 = (const float*)d_in[2];
    const float* Ws2 = (const float*)d_in[3];
    const float* bs2 = (const float*)d_in[4];
    const float* Ws3 = (const float*)d_in[5];
    const float* bs3 = (const float*)d_in[6];
    const float* Wt1 = (const float*)d_in[7];
    const float* bt1 = (const float*)d_in[8];
    const float* Wt2 = (const float*)d_in[9];
    const float* bt2 = (const float*)d_in[10];
    const float* Wt3 = (const float*)d_in[11];
    const float* bt3 = (const float*)d_in[12];

    __bf16* wp = (__bf16*)d_ws;   // 786432 bf16 = 1.5 MB packed weights
    const int S1 = CONDW * HID, S2 = HID * HID, S3 = HID * TOUT;

    pack_w_kernel<<<(S1 + 255) / 256, 256, 0, stream>>>(Ws1, wp,                    CONDW, HID);
    pack_w_kernel<<<(S2 + 255) / 256, 256, 0, stream>>>(Ws2, wp + S1,               HID,   HID);
    pack_w_kernel<<<(S3 + 255) / 256, 256, 0, stream>>>(Ws3, wp + S1 + S2,          HID,  TOUT);
    pack_w_kernel<<<(S1 + 255) / 256, 256, 0, stream>>>(Wt1, wp + S1 + S2 + S3,     CONDW, HID);
    pack_w_kernel<<<(S2 + 255) / 256, 256, 0, stream>>>(Wt2, wp + 2*S1 + S2 + S3,   HID,   HID);
    pack_w_kernel<<<(S3 + 255) / 256, 256, 0, stream>>>(Wt3, wp + 2*S1 + 2*S2 + S3, HID,  TOUT);

    float* y      = (float*)d_out;
    float* logdet = y + (size_t)NROWS * DIM;

    size_t ldsBytes = (size_t)(8 * 16 * H1_STRIDE + 8 * 16 * H2_STRIDE) * sizeof(__bf16);
    realnvp_kernel<<<NROWS / 128, 256, ldsBytes, stream>>>(
        x, bs1, bs2, bs3, bt1, bt2, bt3, wp, y, logdet);
}
